// MultiHeadAttentionLayer_30176440221865
// MI455X (gfx1250) — matrix operand order repaired and checked by reading, not compiled, not run on hardware
//
#include <hip/hip_runtime.h>
#include <hip/hip_bf16.h>

// MI455X / gfx1250 multi-head attention, f16 WMMA with fp32 accumulation.
// B=2, S=2048, D=1024, H=16, DH=64.

typedef __attribute__((ext_vector_type(8)))  _Float16 v8h;
typedef __attribute__((ext_vector_type(16))) _Float16 v16h;
typedef __attribute__((ext_vector_type(8)))  float    v8f;

#define SEQ    2048
#define DMODEL 1024
#define NHEAD  16
#define DHEAD  64
#define NROWS  4096         // B*S
#define BH     32           // B*H

union AFragU { v16h v; v8h h[2]; };

// A-matrix 16x32 f16 fragment (this code's contiguous k order, not the ISA table order): lane l(0-15)+hi half;
// per-lane halfs = [k0+hi*16 .. +15] contiguous, rows m0+l.
__device__ __forceinline__ v16h load_a_frag(const _Float16* base, int ld,
                                            int m0, int k0, int lane) {
  int l = lane & 15, hi = lane >> 4;
  const _Float16* p = base + (size_t)(m0 + l) * ld + k0 + hi * 16;
  AFragU f;
  f.h[0] = *(const v8h*)(p);
  f.h[1] = *(const v8h*)(p + 8);
  return f.v;
}

// B-matrix 32x16 f16 fragment: lane = column n0+l; K 0..15 (lanes 0-15)
// or 16..31 (lanes 16-31) contiguous. `base` stored [n][k] row-major.
__device__ __forceinline__ v16h load_b_frag(const _Float16* base, int ld,
                                            int n0, int k0, int lane) {
  int l = lane & 15, hi = lane >> 4;
  return *(const v16h*)(base + (size_t)(n0 + l) * ld + k0 + hi * 16);
}

__device__ __forceinline__ v8f wmma_f16(v16h a, v16h b, v8f c) {
  return __builtin_amdgcn_wmma_f32_16x16x32_f16(false, a, false, b, (short)0, c,
                                                false, false);
}

// ---------------------------------------------------------------- converts
__global__ void __launch_bounds__(256) cvt_x_kernel(const float* __restrict__ X,
                                                    _Float16* __restrict__ Xh) {
  int i = blockIdx.x * 256 + threadIdx.x;          // 4096*1024 elements
  Xh[i] = (_Float16)X[i];
}

// Wt[mat][h][e][k] = W{mat}[h][k][e]  (3*16*64*1024 elements)
__global__ void __launch_bounds__(256) cvt_w_kernel(const float* __restrict__ Wq,
                                                    const float* __restrict__ Wk,
                                                    const float* __restrict__ Wv,
                                                    _Float16* __restrict__ Wt) {
  int i = blockIdx.x * 256 + threadIdx.x;
  int k   =  i        & 1023;
  int e   = (i >> 10) & 63;
  int h   = (i >> 16) & 15;
  int mat =  i >> 20;
  const float* src = (mat == 0) ? Wq : (mat == 1) ? Wk : Wv;
  Wt[i] = (_Float16)src[((size_t)h * DMODEL + k) * DHEAD + e];
}

// ---------------------------------------------------------------- QKV GEMM
// Wave tile: 32 rows x 64 cols; 8 waves per block SHARE the same column tile
// (same weights -> coalesced in WGP$) and take consecutive 32-row tiles.
// 6144 wave tiles = 768 blocks * 8 waves. K-loop is register double-buffered.
__global__ void __launch_bounds__(256) qkv_kernel(const _Float16* __restrict__ Xh,
                                                  const _Float16* __restrict__ Wt,
                                                  const float* __restrict__ bq,
                                                  const float* __restrict__ bk,
                                                  const float* __restrict__ bv,
                                                  _Float16* __restrict__ Qh,
                                                  _Float16* __restrict__ Kh,
                                                  _Float16* __restrict__ Vt) {
  int lane = threadIdx.x & 31;
  int wid  = threadIdx.x >> 5;
  int nt   = blockIdx.x % 48;                  // column tile (shared in block)
  int mt   = (blockIdx.x / 48) * 8 + wid;      // 128 row tiles of 32
  int mat  = nt / 16, h = nt % 16;
  int m0   = mt * 32;
  const _Float16* wbase = Wt + ((size_t)(mat * NHEAD + h) * DHEAD) * DMODEL;

  v8f acc[2][4] = {};
  v16h a0 = load_a_frag(Xh, DMODEL, m0,      0, lane);
  v16h a1 = load_a_frag(Xh, DMODEL, m0 + 16, 0, lane);
  v16h b0 = load_b_frag(wbase, DMODEL,  0, 0, lane);
  v16h b1 = load_b_frag(wbase, DMODEL, 16, 0, lane);
  v16h b2 = load_b_frag(wbase, DMODEL, 32, 0, lane);
  v16h b3 = load_b_frag(wbase, DMODEL, 48, 0, lane);

  for (int kk = 0; kk < DMODEL; kk += 32) {
    v16h na0, na1, nb0, nb1, nb2, nb3;
    int kn = kk + 32;
    if (kn < DMODEL) {                         // prefetch next K-chunk
      na0 = load_a_frag(Xh, DMODEL, m0,      kn, lane);
      na1 = load_a_frag(Xh, DMODEL, m0 + 16, kn, lane);
      nb0 = load_b_frag(wbase, DMODEL,  0, kn, lane);
      nb1 = load_b_frag(wbase, DMODEL, 16, kn, lane);
      nb2 = load_b_frag(wbase, DMODEL, 32, kn, lane);
      nb3 = load_b_frag(wbase, DMODEL, 48, kn, lane);
    }
    acc[0][0] = wmma_f16(a0, b0, acc[0][0]);
    acc[0][1] = wmma_f16(a0, b1, acc[0][1]);
    acc[0][2] = wmma_f16(a0, b2, acc[0][2]);
    acc[0][3] = wmma_f16(a0, b3, acc[0][3]);
    acc[1][0] = wmma_f16(a1, b0, acc[1][0]);
    acc[1][1] = wmma_f16(a1, b1, acc[1][1]);
    acc[1][2] = wmma_f16(a1, b2, acc[1][2]);
    acc[1][3] = wmma_f16(a1, b3, acc[1][3]);
    if (kn < DMODEL) {
      a0 = na0; a1 = na1;
      b0 = nb0; b1 = nb1; b2 = nb2; b3 = nb3;
    }
  }

  int l = lane & 15, hi = lane >> 4;
  const float* bias = (mat == 0) ? bq : (mat == 1) ? bk : bv;
#pragma unroll
  for (int rt = 0; rt < 2; ++rt) {
#pragma unroll
    for (int j = 0; j < 4; ++j) {
      int e = j * 16 + l;
      float bval = bias[h * DHEAD + e];
#pragma unroll
      for (int r = 0; r < 8; ++r) {
        float val = acc[rt][j][r] + bval;
        int row = m0 + rt * 16 + r + hi * 8;     // global row in [0,4096)
        int b = row >> 11, s = row & (SEQ - 1);
        size_t bh = (size_t)(b * NHEAD + h);
        if (mat == 0)       Qh[(bh * SEQ + s) * DHEAD + e] = (_Float16)val;
        else if (mat == 1)  Kh[(bh * SEQ + s) * DHEAD + e] = (_Float16)val;
        else                Vt[(bh * DHEAD + e) * SEQ + s] = (_Float16)val;
      }
    }
  }
}

// ---------------------------------------------------------------- attention
// One wave per 16-query-row tile of one (b,h); 8 waves in a block share (b,h)
// and stream the same K/V (L1 coalescing). Full-row streaming softmax
// (denominator over ALL keys), causal mask only on the numerator.
// K-tiles are register double-buffered; next-tile loads overlap the softmax VALU.
__global__ void __launch_bounds__(256) attn_kernel(const _Float16* __restrict__ Qh,
                                                   const _Float16* __restrict__ Kh,
                                                   const _Float16* __restrict__ Vt,
                                                   float* __restrict__ out) {
  __shared__ _Float16 lds[8][16 * 40];   // per-wave 16x32 P staging (stride 40 halfs)

  int lane = threadIdx.x & 31;
  int wid  = threadIdx.x >> 5;
  int w    = blockIdx.x * 8 + wid;
  int qt   = w & 127;                    // 128 query tiles
  int bh   = w >> 7;                     // 0..31
  int m0   = qt * 16;
  int l = lane & 15, hi = lane >> 4;

  const _Float16* Qb = Qh + (size_t)bh * SEQ * DHEAD;
  const _Float16* Kb = Kh + (size_t)bh * SEQ * DHEAD;
  const _Float16* Vb = Vt + (size_t)bh * DHEAD * SEQ;

  v16h aq0 = load_a_frag(Qb, DHEAD, m0, 0, lane);
  v16h aq1 = load_a_frag(Qb, DHEAD, m0, 32, lane);

  v8f acc0 = {}, acc1 = {}, acc2 = {}, acc3 = {};
  float mrow[8], lrow[8];
#pragma unroll
  for (int r = 0; r < 8; ++r) { mrow[r] = -1e30f; lrow[r] = 0.f; }

  _Float16* pbuf = lds[wid];

  v16h kb0 = load_b_frag(Kb, DHEAD,  0, 0,  lane);
  v16h kb1 = load_b_frag(Kb, DHEAD,  0, 32, lane);
  v16h kb2 = load_b_frag(Kb, DHEAD, 16, 0,  lane);
  v16h kb3 = load_b_frag(Kb, DHEAD, 16, 32, lane);

  for (int t0 = 0; t0 < SEQ; t0 += 32) {
    int tn = t0 + 32;
    v16h nk0, nk1, nk2, nk3;
    if (tn < SEQ) {                      // prefetch next key tile
      nk0 = load_b_frag(Kb, DHEAD, tn,      0,  lane);
      nk1 = load_b_frag(Kb, DHEAD, tn,      32, lane);
      nk2 = load_b_frag(Kb, DHEAD, tn + 16, 0,  lane);
      nk3 = load_b_frag(Kb, DHEAD, tn + 16, 32, lane);
    }

    v8f s0 = {}, s1 = {};
    s0 = wmma_f16(aq0, kb0, s0);
    s0 = wmma_f16(aq1, kb1, s0);
    s1 = wmma_f16(aq0, kb2, s1);
    s1 = wmma_f16(aq1, kb3, s1);

    float alpha[8];
#pragma unroll
    for (int r = 0; r < 8; ++r) {
      float a0 = s0[r] * 0.125f;         // 1/sqrt(64)
      float a1 = s1[r] * 0.125f;
      float mx = fmaxf(a0, a1);          // row max across 16-lane half-group
      mx = fmaxf(mx, __shfl_xor(mx, 1, 16));
      mx = fmaxf(mx, __shfl_xor(mx, 2, 16));
      mx = fmaxf(mx, __shfl_xor(mx, 4, 16));
      mx = fmaxf(mx, __shfl_xor(mx, 8, 16));
      float mnew = fmaxf(mrow[r], mx);
      float al   = __expf(mrow[r] - mnew);
      float p0   = __expf(a0 - mnew);
      float p1   = __expf(a1 - mnew);
      float ts   = p0 + p1;              // UNMASKED row sum -> full softmax denom
      ts += __shfl_xor(ts, 1, 16);
      ts += __shfl_xor(ts, 2, 16);
      ts += __shfl_xor(ts, 4, 16);
      ts += __shfl_xor(ts, 8, 16);
      lrow[r] = lrow[r] * al + ts;
      mrow[r] = mnew;
      alpha[r] = al;
      s0[r] = p0; s1[r] = p1;
    }
#pragma unroll
    for (int r = 0; r < 8; ++r) {
      acc0[r] *= alpha[r]; acc1[r] *= alpha[r];
      acc2[r] *= alpha[r]; acc3[r] *= alpha[r];
    }

    // Numerator only for chunks intersecting the causal triangle (uniform branch)
    if (t0 <= m0 + 15) {
      // Issue V loads first so they hide under the LDS transpose round-trip.
      v16h vb0 = load_b_frag(Vb, SEQ, 0,  t0, lane);
      v16h vb1 = load_b_frag(Vb, SEQ, 16, t0, lane);
      v16h vb2 = load_b_frag(Vb, SEQ, 32, t0, lane);
      v16h vb3 = load_b_frag(Vb, SEQ, 48, t0, lane);
      // C layout -> A layout transpose through LDS, applying tril mask
#pragma unroll
      for (int r = 0; r < 8; ++r) {
        int qrow = r + hi * 8;
        int qg   = m0 + qrow;
        float p0 = (t0 + l      <= qg) ? s0[r] : 0.f;
        float p1 = (t0 + 16 + l <= qg) ? s1[r] : 0.f;
        pbuf[qrow * 40 + l]      = (_Float16)p0;
        pbuf[qrow * 40 + 16 + l] = (_Float16)p1;
      }
      asm volatile("s_wait_dscnt 0" ::: "memory");
      v16h ap = load_a_frag((const _Float16*)pbuf, 40, 0, 0, lane);
      acc0 = wmma_f16(ap, vb0, acc0);
      acc1 = wmma_f16(ap, vb1, acc1);
      acc2 = wmma_f16(ap, vb2, acc2);
      acc3 = wmma_f16(ap, vb3, acc3);
    }

    if (tn < SEQ) { kb0 = nk0; kb1 = nk1; kb2 = nk2; kb3 = nk3; }
  }

  // out[b][s][h*64+e] = numerator / full-row denominator
  int b = bh >> 4, h = bh & 15;
#pragma unroll
  for (int r = 0; r < 8; ++r) {
    int srow = m0 + r + hi * 8;
    float inv = 1.f / lrow[r];
    size_t base = ((size_t)b * SEQ + srow) * DMODEL + h * DHEAD;
    out[base + 0  + l] = acc0[r] * inv;
    out[base + 16 + l] = acc1[r] * inv;
    out[base + 32 + l] = acc2[r] * inv;
    out[base + 48 + l] = acc3[r] * inv;
  }
}

// ---------------------------------------------------------------- launcher
extern "C" void kernel_launch(void* const* d_in, const int* in_sizes, int n_in,
                              void* d_out, int out_size, void* d_ws, size_t ws_size,
                              hipStream_t stream) {
  const float* X  = (const float*)d_in[0];
  const float* Wq = (const float*)d_in[1];
  const float* bq = (const float*)d_in[2];
  const float* Wk = (const float*)d_in[3];
  const float* bk = (const float*)d_in[4];
  const float* Wv = (const float*)d_in[5];
  const float* bv = (const float*)d_in[6];
  float* out = (float*)d_out;

  char* ws = (char*)d_ws;
  _Float16* Xh = (_Float16*)ws; ws += (size_t)NROWS * DMODEL * 2;             // 8 MB
  _Float16* Wt = (_Float16*)ws; ws += (size_t)3 * NHEAD * DHEAD * DMODEL * 2; // 6 MB
  _Float16* Qh = (_Float16*)ws; ws += (size_t)BH * SEQ * DHEAD * 2;           // 8 MB
  _Float16* Kh = (_Float16*)ws; ws += (size_t)BH * SEQ * DHEAD * 2;           // 8 MB
  _Float16* Vt = (_Float16*)ws;                                               // 8 MB

  cvt_x_kernel<<<(NROWS * DMODEL) / 256, 256, 0, stream>>>(X, Xh);
  cvt_w_kernel<<<(3 * NHEAD * DHEAD * DMODEL) / 256, 256, 0, stream>>>(Wq, Wk, Wv, Wt);
  qkv_kernel<<<768, 256, 0, stream>>>(Xh, Wt, bq, bk, bv, Qh, Kh, Vt);
  attn_kernel<<<512, 256, 0, stream>>>(Qh, Kh, Vt, out);
}
